// MultiHeadAttention_14396730376796
// MI455X (gfx1250) — matrix operand order repaired and checked by reading, not compiled, not run on hardware
//
#include <hip/hip_runtime.h>
#include <hip/hip_bf16.h>

// ---------------------------------------------------------------------------
// MI455X (gfx1250) fused MHA forward.
//  - All matmuls on v_wmma_f32_16x16x32_bf16 (fp32 accumulate).
//  - Attention scores kept in LDS (320KB/WGP); probs written to HBM once.
//  - V tiles double-buffered in LDS via global_load_async_to_lds_b128
//    (ASYNCcnt), overlapping the copy of tile n+1 with WMMAs of tile n.
//  - wave32 everywhere; cross-half reductions via __shfl_xor(...,16).
// ---------------------------------------------------------------------------

#define B_  4
#define S_  1024
#define D_  1024
#define H_  16
#define DK_ 64

typedef __bf16 bf16;
typedef __attribute__((ext_vector_type(16))) __bf16 v16bf;
typedef __attribute__((ext_vector_type(8)))  __bf16 v8bf;
typedef __attribute__((ext_vector_type(8)))  float   v8f;
typedef __attribute__((ext_vector_type(4)))  float   v4f;

__device__ __forceinline__ v8f wmma_bf16(v16bf a, v16bf b, v8f c) {
  // 8 args: (neg_a, A, neg_b, B, c_mod, C, reuse_a, reuse_b)
  return __builtin_amdgcn_wmma_f32_16x16x32_bf16(false, a, false, b,
                                                 (short)0, c, false, false);
}

// ---------------------------------------------------------------------------
// Generic GEMM: Out[M,N] = A[M,K] @ W[N,K]^T + bias[N]
// Workgroup: 256 threads = 8 waves in a 4(M) x 2(N) arrangement.
// Wave tile: 16 rows x 64 cols (4 accumulators). K stepped by 32.
// ---------------------------------------------------------------------------
template <bool A_BF16, bool OUT_F32>
__global__ __launch_bounds__(256) void gemm_bias_kernel(
    const void* __restrict__ Aptr, const float* __restrict__ W,
    const float* __restrict__ bias, void* __restrict__ Out,
    int M, int N, int K) {
  const int tid  = threadIdx.x;
  const int wave = tid >> 5;
  const int lane = tid & 31;
  const int l16  = lane & 15;
  const int hb   = lane >> 4;          // half-wave index (0/1)
  const int wm   = wave & 3;
  const int wn   = wave >> 2;

  const int row0 = blockIdx.x * 64 + wm * 16;
  const int col0 = blockIdx.y * 128 + wn * 64;
  const int arow = row0 + l16;

  v8f acc[4] = {};

  for (int k0 = 0; k0 < K; k0 += 32) {
    // ---- A fragment ----
    v16bf a;
    if (A_BF16) {
      const bf16* Ab = (const bf16*)Aptr + (size_t)arow * K + k0 + hb * 8;
      v8bf lo = *(const v8bf*)Ab;
      v8bf hi = *(const v8bf*)(Ab + 16);
#pragma unroll
      for (int i = 0; i < 8; ++i) { a[i] = lo[i]; a[8 + i] = hi[i]; }
    } else {
      const float* Af = (const float*)Aptr + (size_t)arow * K + k0 + hb * 8;
      v4f f0 = *(const v4f*)(Af +  0);
      v4f f1 = *(const v4f*)(Af +  4);
      v4f f2 = *(const v4f*)(Af + 16);
      v4f f3 = *(const v4f*)(Af + 20);
#pragma unroll
      for (int i = 0; i < 4; ++i) {
        a[i]      = (bf16)f0[i];  a[4 + i]  = (bf16)f1[i];
        a[8 + i]  = (bf16)f2[i];  a[12 + i] = (bf16)f3[i];
      }
    }
    // ---- 4 B fragments + WMMA ----
#pragma unroll
    for (int t = 0; t < 4; ++t) {
      const int n = col0 + t * 16 + l16;
      const float* wr = W + (size_t)n * K + k0 + hb * 8;
      __builtin_prefetch(wr + 32, 0, 1);          // global_prefetch_b8
      v4f w0 = *(const v4f*)(wr + 0);
      v4f w1 = *(const v4f*)(wr + 4);
      v4f w2 = *(const v4f*)(wr + 16);
      v4f w3 = *(const v4f*)(wr + 20);
      v16bf bfrag;
#pragma unroll
      for (int i = 0; i < 4; ++i) {
        bfrag[i]      = (bf16)w0[i];  bfrag[4 + i]  = (bf16)w1[i];
        bfrag[8 + i]  = (bf16)w2[i];  bfrag[12 + i] = (bf16)w3[i];
      }
      acc[t] = wmma_bf16(a, bfrag, acc[t]);
    }
  }

  // ---- epilogue: bias + store (D layout: row = r + 8*hb, col = l16) ----
#pragma unroll
  for (int t = 0; t < 4; ++t) {
    const int n = col0 + t * 16 + l16;
    const float bv = bias[n];
#pragma unroll
    for (int r = 0; r < 8; ++r) {
      const int m = row0 + r + hb * 8;
      const float v = acc[t][r] + bv;
      if (OUT_F32) ((float*)Out)[(size_t)m * N + n] = v;
      else         ((bf16*)Out)[(size_t)m * N + n] = (bf16)v;
    }
  }
}

// ---------------------------------------------------------------------------
// Attention: per block = (b, h, 64 query rows), 4 waves, 16 q-rows per wave.
// LDS layout (dynamic, single allocation, byte offsets known statically):
//   [0, SCORE_BYTES)       : 4 waves x 16 x ROWSTRIDE fp32 score strips
//   [VBUF_OFF, +32KB)      : 4 waves x 2 x (32x64 bf16) V-tile double buffers
// ---------------------------------------------------------------------------
#define ROWSTRIDE 1033                   // 1024+9: stride%64==9, conflict-free
#define SCORE_BYTES (4 * 16 * ROWSTRIDE * 4)       // 264448
#define VBUF_OFF    SCORE_BYTES                    // 16B aligned
#define VTILE_ELEMS (32 * 64)                      // bf16 elems per tile (4KB)
#define VROWB       128                            // bytes per V-tile row
#define SMEM_BYTES  (VBUF_OFF + 4 * 2 * VTILE_ELEMS * 2)   // 297216

// Issue one 32x64-bf16 V tile copy (4KB) as 8 async 512B global->LDS ops.
// Wave-private buffer, tracked by ASYNCcnt; no workgroup barrier required.
__device__ __forceinline__ void v_tile_prefetch(const bf16* __restrict__ Vb,
                                                int b, int h, int n0,
                                                int wave, int buf, int lane) {
  const int sub  = lane >> 3;                       // 0..3: row within group
  const int colb = (lane & 7) * 16;                 // byte column (8x16B=128B)
  const unsigned lds0 =
      (unsigned)(VBUF_OFF + (wave * 2 + buf) * (VTILE_ELEMS * 2) + colb);
  const char* g0 =
      (const char*)(Vb + ((size_t)(b * S_) + n0) * D_ + h * DK_) + colb;
#pragma unroll
  for (int i = 0; i < 8; ++i) {
    const int row = i * 4 + sub;                    // 0..31 key rows
    unsigned lds = lds0 + (unsigned)(row * VROWB);
    unsigned long long ga = (unsigned long long)(g0 + (size_t)row * D_ * 2);
    asm volatile("global_load_async_to_lds_b128 %0, %1, off"
                 :: "v"(lds), "v"(ga) : "memory");
  }
}

__global__ __launch_bounds__(128) void attn_kernel(
    const bf16* __restrict__ Qb, const bf16* __restrict__ Kb,
    const bf16* __restrict__ Vb, const int* __restrict__ mask,
    float* __restrict__ attn_out, bf16* __restrict__ ctx) {
  extern __shared__ float smem[];

  const int tid  = threadIdx.x;
  const int wave = tid >> 5;
  const int lane = tid & 31;
  const int l16  = lane & 15;
  const int hb   = lane >> 4;

  const int qt = blockIdx.x, h = blockIdx.y, b = blockIdx.z;
  const int qbase = qt * 64 + wave * 16;
  float* myrow = smem + (size_t)wave * 16 * ROWSTRIDE;

  // ---- Q fragments (K = 64 -> two 16x32 fragments), loaded once ----
  v16bf qf[2];
  {
    const bf16* qp = Qb + ((size_t)(b * S_) + qbase + l16) * D_ + h * DK_;
#pragma unroll
    for (int j = 0; j < 2; ++j) {
      const bf16* p = qp + j * 32 + hb * 8;
      v8bf lo = *(const v8bf*)p;
      v8bf hi = *(const v8bf*)(p + 16);
#pragma unroll
      for (int i = 0; i < 8; ++i) { qf[j][i] = lo[i]; qf[j][8 + i] = hi[i]; }
    }
  }

  // ---- Phase 1: scores = QK^T * 0.125 (+mask) -> LDS ----
  for (int n0 = 0; n0 < S_; n0 += 16) {
    v8f s = {};
#pragma unroll
    for (int j = 0; j < 2; ++j) {
      const bf16* kp = Kb + ((size_t)(b * S_) + n0 + l16) * D_ + h * DK_
                          + j * 32 + hb * 8;
      v8bf lo = *(const v8bf*)kp;
      v8bf hi = *(const v8bf*)(kp + 16);
      v16bf kf;
#pragma unroll
      for (int i = 0; i < 8; ++i) { kf[i] = lo[i]; kf[8 + i] = hi[i]; }
      s = wmma_bf16(qf[j], kf, s);
    }
    const int ncol = n0 + l16;
    const int mk = mask[b * S_ + ncol];
#pragma unroll
    for (int r = 0; r < 8; ++r) {
      const float v = mk ? s[r] * 0.125f : -1e9f;
      myrow[(r + hb * 8) * ROWSTRIDE + ncol] = v;   // ds_store
    }
  }

  // ---- Softmax over each wave-private row (DS ops in-order per wave) ----
  {
    float* rowp = myrow + l16 * ROWSTRIDE + hb * 512;
    float mx = -3.0e38f;
    for (int c = 0; c < 512; ++c) mx = fmaxf(mx, rowp[c]);
    mx = fmaxf(mx, __shfl_xor(mx, 16, 32));
    float sm = 0.f;
    for (int c = 0; c < 512; ++c) {
      float e = __expf(rowp[c] - mx);
      rowp[c] = e;
      sm += e;
    }
    sm += __shfl_xor(sm, 16, 32);
    const float inv = 1.0f / sm;
    float* aout = attn_out + ((size_t)(b * H_ + h) * S_ + qbase + l16) * S_
                           + hb * 512;
    for (int c = 0; c < 512; ++c) {
      const float p = rowp[c] * inv;
      rowp[c] = p;           // probs now live in LDS for the PV matmul
      aout[c] = p;           // required output: full attention tensor
    }
  }

  // ---- Phase 2: ctx = P @ V with async-LDS double-buffered V tiles ----
  bf16* vball = (bf16*)((char*)smem + VBUF_OFF);
  v_tile_prefetch(Vb, b, h, /*n0=*/0, wave, /*buf=*/0, lane);

  v8f oacc[4] = {};
  for (int n0 = 0; n0 < S_; n0 += 32) {
    const int cur = (n0 >> 5) & 1;
    // current V tile landed in LDS
    asm volatile("s_wait_asynccnt 0x0" ::: "memory");

    // P fragment (A-layout) from LDS scores, cvt fp32->bf16
    v16bf pf;
    const float* pr = myrow + l16 * ROWSTRIDE + n0 + hb * 8;
#pragma unroll
    for (int i = 0; i < 8; ++i) {
      pf[i]     = (bf16)pr[i];
      pf[8 + i] = (bf16)pr[16 + i];
    }

    // V fragments (B-layout) gathered from the staged LDS tile
    const bf16* vt = vball + (wave * 2 + cur) * VTILE_ELEMS;
    v16bf vf[4];
#pragma unroll
    for (int dt = 0; dt < 4; ++dt)
#pragma unroll
      for (int i = 0; i < 16; ++i)
        vf[dt][i] = vt[(hb * 8 + i + (i >> 3) * 8) * 64 + dt * 16 + l16];

    // all DS reads of the other buffer retired -> safe to overwrite it
    asm volatile("s_wait_dscnt 0x0" ::: "memory");
    if (n0 + 32 < S_)
      v_tile_prefetch(Vb, b, h, n0 + 32, wave, cur ^ 1, lane);

#pragma unroll
    for (int dt = 0; dt < 4; ++dt)
      oacc[dt] = wmma_bf16(pf, vf[dt], oacc[dt]);
  }

  // ---- Phase 3: bf16 context out (merge-heads layout [B,S,H*DK]) ----
#pragma unroll
  for (int dt = 0; dt < 4; ++dt) {
    const int col = h * DK_ + dt * 16 + l16;
#pragma unroll
    for (int r = 0; r < 8; ++r) {
      const int m = qbase + r + hb * 8;
      ctx[((size_t)(b * S_) + m) * D_ + col] = (bf16)oacc[dt][r];
    }
  }
}

// ---------------------------------------------------------------------------
extern "C" void kernel_launch(void* const* d_in, const int* in_sizes, int n_in,
                              void* d_out, int out_size, void* d_ws,
                              size_t ws_size, hipStream_t stream) {
  const float* q    = (const float*)d_in[0];
  const float* k    = (const float*)d_in[1];
  const float* v    = (const float*)d_in[2];
  const int*   mask = (const int*)d_in[3];
  const float* Wq = (const float*)d_in[4];  const float* bq = (const float*)d_in[5];
  const float* Wk = (const float*)d_in[6];  const float* bk = (const float*)d_in[7];
  const float* Wv = (const float*)d_in[8];  const float* bv = (const float*)d_in[9];
  const float* Wo = (const float*)d_in[10]; const float* bo = (const float*)d_in[11];

  float* out  = (float*)d_out;                       // [B,S,D]
  float* attn = out + (size_t)B_ * S_ * D_;          // [B,H,S,S]

  const size_t nelem = (size_t)B_ * S_ * D_;         // 4M elements
  bf16* Qb  = (bf16*)d_ws;
  bf16* Kb  = Qb + nelem;
  bf16* Vb  = Kb + nelem;
  bf16* Ctx = Vb + nelem;                            // 32 MB total scratch

  const int M = B_ * S_;                             // 4096
  dim3 gemm_grid(M / 64, D_ / 128);                  // (64, 8)

  // Q/K/V projections: fp32 in -> bf16 out (feeds attention WMMA directly)
  gemm_bias_kernel<false, false><<<gemm_grid, 256, 0, stream>>>(q, Wq, bq, Qb, M, D_, D_);
  gemm_bias_kernel<false, false><<<gemm_grid, 256, 0, stream>>>(k, Wk, bk, Kb, M, D_, D_);
  gemm_bias_kernel<false, false><<<gemm_grid, 256, 0, stream>>>(v, Wv, bv, Vb, M, D_, D_);

  // Attention core: scores/softmax in LDS, probs -> d_out, ctx -> scratch
  attn_kernel<<<dim3(S_ / 64, H_, B_), 128, SMEM_BYTES, stream>>>(
      Qb, Kb, Vb, mask, attn, Ctx);

  // Output projection: bf16 in -> fp32 out
  gemm_bias_kernel<true, true><<<gemm_grid, 256, 0, stream>>>(Ctx, Wo, bo, out, M, D_, D_);
}
